// CrossAttention_74818330296332
// MI455X (gfx1250) — matrix operand order repaired and checked by reading, not compiled, not run on hardware
//
#include <hip/hip_runtime.h>
#include <hip/hip_bf16.h>

typedef __attribute__((ext_vector_type(16))) __bf16 v16bf;
typedef __attribute__((ext_vector_type(8)))  __bf16 v8bf;
typedef __attribute__((ext_vector_type(4)))  __bf16 v4bf;
typedef __bf16 v2bf __attribute__((ext_vector_type(2), aligned(4)));
typedef __attribute__((ext_vector_type(8)))  float  v8f;
typedef __attribute__((ext_vector_type(4)))  float  v4f;
typedef float v4fu __attribute__((ext_vector_type(4), aligned(4)));  // DWORD-aligned vec load

#define NB   8
#define NC   128
#define NCC  256
#define NH   256
#define NW   256
#define NHW  (NH*NW)
#define KCONV 2304          // 256*9
#define LDB  40             // K-pitch (elements) of transposed B tiles: 80B, 16B-multiple

static __device__ inline v8f vzero() {
  v8f z = {0.f,0.f,0.f,0.f,0.f,0.f,0.f,0.f};
  return z;
}

// ---- CDNA5 WMMA fragment builders (wave32 layouts, CDNA5 ISA) ----
static __device__ inline v16bf ld_a(const __bf16* s, int ld) {
  int lane = threadIdx.x & 31;
  int m  = lane & 15;
  int hi = (lane >> 4) * 8;
  const __bf16* base = s + m*ld + hi;
  v8bf lo = *(const v8bf*)(base);
  v8bf hh = *(const v8bf*)(base + 16);
  return __builtin_shufflevector(lo, hh, 0,1,2,3,4,5,6,7,8,9,10,11,12,13,14,15);
}
static __device__ inline v16bf ld_a_f32(const float* s, int ld) {
  int lane = threadIdx.x & 31;
  int m  = lane & 15;
  int hi = (lane >> 4) * 8;
  const float* base = s + m*ld + hi;
  v8f lo = *(const v8f*)(base);
  v8f hh = *(const v8f*)(base + 16);
  v16bf a;
#pragma unroll
  for (int e = 0; e < 8; ++e) { a[e] = (__bf16)lo[e]; a[e+8] = (__bf16)hh[e]; }
  return a;
}
static __device__ inline v16bf ld_bt(const __bf16* sT) {
  int lane = threadIdx.x & 31;
  int n  = lane & 15;
  int kb = (lane >> 4) * 8;
  const __bf16* base = sT + n*LDB + kb;
  v8bf lo = *(const v8bf*)(base);
  v8bf hh = *(const v8bf*)(base + 16);
  return __builtin_shufflevector(lo, hh, 0,1,2,3,4,5,6,7,8,9,10,11,12,13,14,15);
}
// D frag -> LDS tile (rows = (lane>>4)*8 + r, col = lane&15)
static __device__ inline void frag_to_lds_bf16(__bf16* sO, int ld, int row0, int col0, v8f d) {
  int lane = threadIdx.x & 31;
  int col = col0 + (lane & 15), rb = row0 + (lane >> 4) * 8;
#pragma unroll
  for (int r = 0; r < 8; ++r) sO[(rb + r)*ld + col] = (__bf16)d[r];
}
static __device__ inline void store_d_f32s(float* s, int ld, v8f d, float scale) {
  int lane = threadIdx.x & 31;
  int col = lane & 15, rb = (lane >> 4) * 8;
#pragma unroll
  for (int r = 0; r < 8; ++r) s[(rb + r)*ld + col] = d[r] * scale;
}

#define WMMA_BF16(A,B,Cacc) \
  __builtin_amdgcn_wmma_f32_16x16x32_bf16(false, (A), false, (B), (short)0, (Cacc), false, false)

// Stage f32 x-tile [32K x 128N] into transposed bf16 [N][LDB]; src points at (k0,p0).
// Two K-rows per thread-chunk -> full v_cvt_pk_bf16_f32 + packed ds_store_b32.
static __device__ inline void stage_x(const float* __restrict__ src, __bf16* sBt, int tid) {
  for (int idx = tid; idx < 512; idx += 256) {       // kk-pair x j4
    int kkp = idx >> 5, j4 = (idx & 31) * 4;
    const float* s0 = src + (size_t)(2*kkp)*NHW + j4;
    v4f a0 = *(const v4f*)(s0);                      // global_load_b128
    v4f a1 = *(const v4f*)(s0 + NHW);                // global_load_b128
#pragma unroll
    for (int u = 0; u < 4; ++u) {
      v2bf pv; pv[0] = (__bf16)a0[u]; pv[1] = (__bf16)a1[u];
      *(v2bf*)(sBt + (j4 + u)*LDB + 2*kkp) = pv;     // ds_store_b32
    }
  }
}
// Stage bf16 tile (attended) the same way (pairs of rows, packed b32 stores).
static __device__ inline void stage_att(const __bf16* __restrict__ src, __bf16* sBt, int tid) {
  {
    int idx = tid;                                    // exactly 256 chunks
    int kkp = idx >> 4, j8 = (idx & 15) * 8;
    const __bf16* s0 = src + (size_t)(2*kkp)*NHW + j8;
    v8bf a0 = *(const v8bf*)(s0);
    v8bf a1 = *(const v8bf*)(s0 + NHW);
#pragma unroll
    for (int u = 0; u < 8; ++u) {
      v2bf pv; pv[0] = a0[u]; pv[1] = a1[u];
      *(v2bf*)(sBt + (j8 + u)*LDB + 2*kkp) = pv;
    }
  }
}

// ============ Kernel 1: q = Wq (1x1) @ x, output bf16 ============
__global__ __launch_bounds__(256) void qproj_kernel(
    const float* __restrict__ x, const float* __restrict__ Wq,
    __bf16* __restrict__ qws) {
  __shared__ __align__(32) unsigned char smem[53248];
  __bf16* sW   = (__bf16*)smem;                 // 32KB
  __bf16* sBt0 = (__bf16*)(smem + 32768);       // 10KB
  __bf16* sBt1 = (__bf16*)(smem + 43008);       // 10KB
  const int tid = threadIdx.x;
  const int b  = blockIdx.x >> 9;
  const int p0 = (blockIdx.x & 511) << 7;
  for (int idx = tid; idx < NC*32; idx += 256) {       // Wq: vector4 both sides
    int m = idx >> 5, k4 = (idx & 31) * 4;
    v4f wv = *(const v4f*)(Wq + m*NC + k4);
    v4bf cv;
#pragma unroll
    for (int u = 0; u < 4; ++u) cv[u] = (__bf16)wv[u];
    *(v4bf*)(sW + m*NC + k4) = cv;
  }
  const int wid = tid >> 5, wm = wid & 1, wn = wid >> 1;
  v8f acc[4][2];
#pragma unroll
  for (int fm = 0; fm < 4; ++fm)
#pragma unroll
    for (int fn = 0; fn < 2; ++fn) acc[fm][fn] = vzero();

  const float* xsrc = x + (size_t)b*NC*NHW + p0;
  stage_x(xsrc, sBt0, tid);
  for (int it = 0; it < 4; ++it) {
    __syncthreads();
    __bf16* sBcur = (it & 1) ? sBt1 : sBt0;
    if (it + 1 < 4) stage_x(xsrc + (size_t)(it + 1)*32*NHW, (it & 1) ? sBt0 : sBt1, tid);
#pragma unroll
    for (int fm = 0; fm < 4; ++fm) {
      v16bf a = ld_a(sW + (64*wm + 16*fm)*NC + it*32, NC);
#pragma unroll
      for (int fn = 0; fn < 2; ++fn) {
        v16bf bb = ld_bt(sBcur + (32*wn + 16*fn)*LDB);
        acc[fm][fn] = WMMA_BF16(a, bb, acc[fm][fn]);
      }
    }
  }
  // epilogue: frags -> LDS -> coalesced b128 stores
  __syncthreads();
  __bf16* sO = (__bf16*)smem;                   // reuse (tiles dead)
#pragma unroll
  for (int fm = 0; fm < 4; ++fm)
#pragma unroll
    for (int fn = 0; fn < 2; ++fn)
      frag_to_lds_bf16(sO, 128, 64*wm + 16*fm, 32*wn + 16*fn, acc[fm][fn]);
  __syncthreads();
  for (int idx = tid; idx < 128*16; idx += 256) {
    int row = idx >> 4, c8 = (idx & 15) * 8;
    *(v8bf*)(qws + ((size_t)b*NC + row)*NHW + p0 + c8) = *(const v8bf*)(sO + row*128 + c8);
  }
}

// ===== Kernel 2: kv = conv3x3(cond, Wkv)+bkv as implicit GEMM, split k/v =====
static __device__ inline void stage_conv(const float* __restrict__ Wkv,
                                         const float* __restrict__ cond,
                                         int tid, int b, int m0, int k0, int y, int x0,
                                         __bf16* sA, __bf16* sBt) {
  for (int idx = tid; idx < 128*8; idx += 256) {       // weights: vector4 both sides
    int im = idx >> 3, k4 = (idx & 7) * 4;
    v4f wv = *(const v4f*)(Wkv + (size_t)(m0 + im)*KCONV + k0 + k4);
    v4bf cv;
#pragma unroll
    for (int u = 0; u < 4; ++u) cv[u] = (__bf16)wv[u];
    *(v4bf*)(sA + im*32 + k4) = cv;
  }
  // im2col gather: two k-rows per chunk -> packed b32 LDS stores
  for (int idx = tid; idx < 512; idx += 256) {
    int kkp = idx >> 5, j4 = (idx & 31) * 4;
    float f[2][4];
#pragma unroll
    for (int h = 0; h < 2; ++h) {
      int k  = k0 + 2*kkp + h;
      int ci = k / 9, r9 = k - ci*9;
      int ky = r9 / 3, kx = r9 - ky*3;
      int yy = y + ky - 1;
      int xb = x0 + j4 + kx - 1;
      f[h][0] = 0.f; f[h][1] = 0.f; f[h][2] = 0.f; f[h][3] = 0.f;
      if ((unsigned)yy < (unsigned)NH) {
        const float* src = cond + ((size_t)b*NCC + ci)*NHW + (size_t)yy*NW;
        if (xb >= 0 && xb + 3 < NW) {
          v4fu cv = *(const v4fu*)(src + xb);
#pragma unroll
          for (int u = 0; u < 4; ++u) f[h][u] = cv[u];
        } else {
#pragma unroll
          for (int u = 0; u < 4; ++u) {
            int xx = xb + u;
            if ((unsigned)xx < (unsigned)NW) f[h][u] = src[xx];
          }
        }
      }
    }
#pragma unroll
    for (int u = 0; u < 4; ++u) {
      v2bf pv; pv[0] = (__bf16)f[0][u]; pv[1] = (__bf16)f[1][u];
      *(v2bf*)(sBt + (j4 + u)*LDB + 2*kkp) = pv;
    }
  }
}

__global__ __launch_bounds__(256) void kvconv_kernel(
    const float* __restrict__ cond, const float* __restrict__ Wkv,
    const float* __restrict__ bkv,
    __bf16* __restrict__ kws, __bf16* __restrict__ vws) {
  __shared__ __align__(32) unsigned char smem[37376];
  __bf16* sA0  = (__bf16*)smem;                 // 8KB
  __bf16* sA1  = (__bf16*)(smem + 8192);
  __bf16* sBt0 = (__bf16*)(smem + 16384);       // 10KB
  __bf16* sBt1 = (__bf16*)(smem + 26624);
  float*  sBias = (float*)(smem + 36864);       // 512B
  const int tid = threadIdx.x;
  const int b   = blockIdx.x >> 10;
  const int rem = blockIdx.x & 1023;
  const int m0  = (rem >> 9) << 7;              // 0 or 128
  const int p0  = (rem & 511) << 7;
  const int y   = p0 >> 8;
  const int x0  = p0 & 255;
  if (tid < 128) sBias[tid] = bkv[m0 + tid];
  const int wid = tid >> 5, wm = wid & 1, wn = wid >> 1;
  v8f acc[4][2];
#pragma unroll
  for (int fm = 0; fm < 4; ++fm)
#pragma unroll
    for (int fn = 0; fn < 2; ++fn) acc[fm][fn] = vzero();

  stage_conv(Wkv, cond, tid, b, m0, 0, y, x0, sA0, sBt0);
  const int NIT = KCONV / 32;                   // 72
  for (int it = 0; it < NIT; ++it) {
    __syncthreads();
    __bf16* sAcur = (it & 1) ? sA1 : sA0;
    __bf16* sBcur = (it & 1) ? sBt1 : sBt0;
    if (it + 1 < NIT) {
      __builtin_prefetch(&Wkv[(size_t)(m0 + (tid >> 1))*KCONV + (it + 1)*32 + (tid & 1)*16], 0, 1);
      stage_conv(Wkv, cond, tid, b, m0, (it + 1)*32, y, x0,
                 (it & 1) ? sA0 : sA1, (it & 1) ? sBt0 : sBt1);
    }
#pragma unroll
    for (int fm = 0; fm < 4; ++fm) {
      v16bf a = ld_a(sAcur + (64*wm + 16*fm)*32, 32);
#pragma unroll
      for (int fn = 0; fn < 2; ++fn) {
        v16bf bb = ld_bt(sBcur + (32*wn + 16*fn)*LDB);
        acc[fm][fn] = WMMA_BF16(a, bb, acc[fm][fn]);
      }
    }
  }
  // epilogue: bias, frags -> LDS -> coalesced stores into k or v plane
  __syncthreads();
  __bf16* sO = (__bf16*)smem;                   // 32KB, tiles dead
  {
    const int lane = tid & 31, col = lane & 15, rb0 = (lane >> 4) * 8;
#pragma unroll
    for (int fm = 0; fm < 4; ++fm)
#pragma unroll
      for (int fn = 0; fn < 2; ++fn) {
        int r0 = 64*wm + 16*fm, c0 = 32*wn + 16*fn + col;
#pragma unroll
        for (int r = 0; r < 8; ++r) {
          int row = r0 + rb0 + r;
          sO[row*128 + c0] = (__bf16)(acc[fm][fn][r] + sBias[row]);
        }
      }
  }
  __syncthreads();
  __bf16* dst = (m0 == 0) ? kws : vws;
  for (int idx = tid; idx < 128*16; idx += 256) {
    int row = idx >> 4, c8 = (idx & 15) * 8;
    *(v8bf*)(dst + ((size_t)b*NC + row)*NHW + p0 + c8) = *(const v8bf*)(sO + row*128 + c8);
  }
}

// ===== Kernel 3: per-(b,c) row attention; 32 query rows per block =====
__global__ __launch_bounds__(256) void attn_kernel(
    const __bf16* __restrict__ qws, const __bf16* __restrict__ kws,
    const __bf16* __restrict__ vws, __bf16* __restrict__ ows) {
  __shared__ __align__(32) unsigned char smem[56320];
  float*  sS   = (float*)smem;                  // 32KB scores/probs
  __bf16* sT   = (__bf16*)(smem + 32768);       // 20KB transposed K/V tile
  __bf16* sQ   = (__bf16*)(smem + 53248);       // 2KB
  float*  sRed = (float*)(smem + 55296);        // 1KB
  const int tid = threadIdx.x;
  const int rt = blockIdx.x & 7;
  const int c  = (blockIdx.x >> 3) & 127;
  const int b  = blockIdx.x >> 10;
  const int m0 = rt << 5;
  const __bf16* q  = qws + ((size_t)b*NC + c)*NHW;
  const __bf16* kk = kws + ((size_t)b*NC + c)*NHW;
  const __bf16* vv = vws + ((size_t)b*NC + c)*NHW;
  __bf16*       oo = ows + ((size_t)b*NC + c)*NHW;
  const int wid = tid >> 5;

  // ---- pass 1: S = q @ k^T (K over w) ----
  v8f acc[2][2];
#pragma unroll
  for (int fm = 0; fm < 2; ++fm)
#pragma unroll
    for (int fn = 0; fn < 2; ++fn) acc[fm][fn] = vzero();
  for (int w0 = 0; w0 < NW; w0 += 32) {
    __syncthreads();
    if (tid < 128) {                             // q chunk: b128 both sides
      int r = tid >> 2, e8 = (tid & 3) * 8;
      *(v8bf*)(sQ + r*32 + e8) = *(const v8bf*)(q + (size_t)(m0 + r)*NW + w0 + e8);
    }
    for (int idx = tid; idx < 1024; idx += 256) { // k tile: b128 load + ds_store_b128
      int g = idx >> 2, e8 = (idx & 3) * 8;
      *(v8bf*)(sT + g*LDB + e8) = *(const v8bf*)(kk + (size_t)g*NW + w0 + e8);
    }
    __syncthreads();
#pragma unroll
    for (int fm = 0; fm < 2; ++fm) {
      v16bf a = ld_a(sQ + 16*fm*32, 32);
#pragma unroll
      for (int fn = 0; fn < 2; ++fn) {
        v16bf bb = ld_bt(sT + (32*wid + 16*fn)*LDB);
        acc[fm][fn] = WMMA_BF16(a, bb, acc[fm][fn]);
      }
    }
  }
  __syncthreads();
  const float scale = 0.08838834764831843f;      // 128^-0.5
#pragma unroll
  for (int fm = 0; fm < 2; ++fm)
#pragma unroll
    for (int fn = 0; fn < 2; ++fn)
      store_d_f32s(sS + 16*fm*256 + 32*wid + 16*fn, 256, acc[fm][fn], scale);
  __syncthreads();

  // ---- softmax over g: 8 threads per row ----
  {
    const int row = tid & 31, part = tid >> 5;
    float* prow = sS + row*256 + part*32;
    float mx = -3.402823466e38f;
#pragma unroll
    for (int e = 0; e < 32; ++e) mx = fmaxf(mx, prow[e]);
    sRed[row*8 + part] = mx;
    __syncthreads();
    float m8 = -3.402823466e38f;
#pragma unroll
    for (int p = 0; p < 8; ++p) m8 = fmaxf(m8, sRed[row*8 + p]);
    __syncthreads();
    float sum = 0.f;
#pragma unroll
    for (int e = 0; e < 32; ++e) { float ev = __expf(prow[e] - m8); prow[e] = ev; sum += ev; }
    sRed[row*8 + part] = sum;
    __syncthreads();
    float s8 = 0.f;
#pragma unroll
    for (int p = 0; p < 8; ++p) s8 += sRed[row*8 + p];
    float inv = 1.f / s8;
#pragma unroll
    for (int e = 0; e < 32; ++e) prow[e] *= inv;
  }
  __syncthreads();

  // ---- pass 2: O = P @ V (K over g) ----
  v8f oacc[2][2];
#pragma unroll
  for (int fm = 0; fm < 2; ++fm)
#pragma unroll
    for (int fn = 0; fn < 2; ++fn) oacc[fm][fn] = vzero();
  for (int g0 = 0; g0 < NH; g0 += 32) {
    __syncthreads();
    for (int idx = tid; idx < 512; idx += 256) { // v tile: g-row pairs, packed b32 stores
      int ggp = idx >> 5, w8 = (idx & 31) * 8;
      const __bf16* s0 = vv + (size_t)(g0 + 2*ggp)*NW + w8;
      v8bf a0 = *(const v8bf*)(s0);
      v8bf a1 = *(const v8bf*)(s0 + NW);
#pragma unroll
      for (int u = 0; u < 8; ++u) {
        v2bf pv; pv[0] = a0[u]; pv[1] = a1[u];
        *(v2bf*)(sT + (w8 + u)*LDB + 2*ggp) = pv;
      }
    }
    __syncthreads();
#pragma unroll
    for (int fm = 0; fm < 2; ++fm) {
      v16bf a = ld_a_f32(sS + 16*fm*256 + g0, 256);
#pragma unroll
      for (int fn = 0; fn < 2; ++fn) {
        v16bf bb = ld_bt(sT + (32*wid + 16*fn)*LDB);
        oacc[fm][fn] = WMMA_BF16(a, bb, oacc[fm][fn]);
      }
    }
  }
  // epilogue via LDS for coalesced b128 stores
  __syncthreads();
  __bf16* sO = (__bf16*)(smem + 32768);          // reuse sT region
#pragma unroll
  for (int fm = 0; fm < 2; ++fm)
#pragma unroll
    for (int fn = 0; fn < 2; ++fn)
      frag_to_lds_bf16(sO, 256, 16*fm, 32*wid + 16*fn, oacc[fm][fn]);
  __syncthreads();
  for (int idx = tid; idx < 1024; idx += 256) {
    int row = idx >> 5, w8 = (idx & 31) * 8;
    *(v8bf*)(oo + (size_t)(m0 + row)*NW + w8) = *(const v8bf*)(sO + row*256 + w8);
  }
}

// ===== Kernel 4: out = Wproj (1x1) @ attended + bproj + x, f32 =====
__global__ __launch_bounds__(256) void proj_kernel(
    const __bf16* __restrict__ att, const float* __restrict__ Wp,
    const float* __restrict__ bp, const float* __restrict__ xin,
    float* __restrict__ out) {
  __shared__ __align__(32) unsigned char smem[53760];
  __bf16* sW   = (__bf16*)smem;                 // 32KB
  __bf16* sBt0 = (__bf16*)(smem + 32768);
  __bf16* sBt1 = (__bf16*)(smem + 43008);
  float*  sBias = (float*)(smem + 53248);       // 512B
  const int tid = threadIdx.x;
  const int b  = blockIdx.x >> 9;
  const int p0 = (blockIdx.x & 511) << 7;
  if (tid < 128) sBias[tid] = bp[tid];
  for (int idx = tid; idx < NC*32; idx += 256) {
    int m = idx >> 5, k4 = (idx & 31) * 4;
    v4f wv = *(const v4f*)(Wp + m*NC + k4);
    v4bf cv;
#pragma unroll
    for (int u = 0; u < 4; ++u) cv[u] = (__bf16)wv[u];
    *(v4bf*)(sW + m*NC + k4) = cv;
  }
  const int wid = tid >> 5, wm = wid & 1, wn = wid >> 1;
  v8f acc[4][2];
#pragma unroll
  for (int fm = 0; fm < 4; ++fm)
#pragma unroll
    for (int fn = 0; fn < 2; ++fn) acc[fm][fn] = vzero();

  const __bf16* asrc = att + (size_t)b*NC*NHW + p0;
  stage_att(asrc, sBt0, tid);
  for (int it = 0; it < 4; ++it) {
    __syncthreads();
    __bf16* sBcur = (it & 1) ? sBt1 : sBt0;
    if (it + 1 < 4) stage_att(asrc + (size_t)(it + 1)*32*NHW, (it & 1) ? sBt0 : sBt1, tid);
#pragma unroll
    for (int fm = 0; fm < 4; ++fm) {
      v16bf a = ld_a(sW + (64*wm + 16*fm)*NC + it*32, NC);
#pragma unroll
      for (int fn = 0; fn < 2; ++fn) {
        v16bf bb = ld_bt(sBcur + (32*wn + 16*fn)*LDB);
        acc[fm][fn] = WMMA_BF16(a, bb, acc[fm][fn]);
      }
    }
  }
  // epilogue in two 64-row halves (f32 tile, 32KB), fused bias + residual
  float* sO = (float*)smem;                     // reuse (tiles dead after sync)
  for (int half = 0; half < 2; ++half) {
    __syncthreads();
    if (wm == half) {
      const int lane = tid & 31, col = lane & 15, rb0 = (lane >> 4) * 8;
#pragma unroll
      for (int fm = 0; fm < 4; ++fm)
#pragma unroll
        for (int fn = 0; fn < 2; ++fn) {
          int r0 = 16*fm, c0 = 32*wn + 16*fn + col;
#pragma unroll
          for (int r = 0; r < 8; ++r) {
            int row = r0 + rb0 + r;
            sO[row*128 + c0] = acc[fm][fn][r] + sBias[64*half + row];
          }
        }
    }
    __syncthreads();
    for (int idx = tid; idx < 64*32; idx += 256) {
      int row = idx >> 5, c4 = (idx & 31) * 4;
      size_t gi = ((size_t)b*NC + 64*half + row)*NHW + p0 + c4;
      v4f ov = *(const v4f*)(sO + row*128 + c4);
      v4f xv = *(const v4f*)(xin + gi);
      *(v4f*)(out + gi) = ov + xv;
    }
  }
}

extern "C" void kernel_launch(void* const* d_in, const int* in_sizes, int n_in,
                              void* d_out, int out_size, void* d_ws, size_t ws_size,
                              hipStream_t stream) {
  const float* x    = (const float*)d_in[0];
  const float* cond = (const float*)d_in[1];
  const float* Wq   = (const float*)d_in[2];
  const float* Wkv  = (const float*)d_in[3];
  const float* bkv  = (const float*)d_in[4];
  const float* Wp   = (const float*)d_in[5];
  const float* bp   = (const float*)d_in[6];
  float* out = (float*)d_out;

  char* ws = (char*)d_ws;
  const size_t plane = (size_t)NB * NC * NHW * sizeof(__bf16);  // 128 MiB
  __bf16* qws = (__bf16*)(ws);
  __bf16* kws = (__bf16*)(ws + plane);
  __bf16* vws = (__bf16*)(ws + 2*plane);
  __bf16* aws = qws;  // attended aliases q: each attn block only reads its own q rows

  qproj_kernel <<<NB*512,  256, 0, stream>>>(x, Wq, qws);
  kvconv_kernel<<<NB*1024, 256, 0, stream>>>(cond, Wkv, bkv, kws, vws);
  attn_kernel  <<<NB*NC*8, 256, 0, stream>>>(qws, kws, vws, aws);
  proj_kernel  <<<NB*512,  256, 0, stream>>>(aws, Wp, bp, x, out);
}
